// SparseMultiHeadAttention4D_4080218931340
// MI455X (gfx1250) — matrix operand order repaired and checked by reading, not compiled, not run on hardware
//
#include <hip/hip_runtime.h>
#include <hip/hip_bf16.h>
#include <math.h>

typedef __attribute__((ext_vector_type(16))) __bf16 v16bf;
typedef __attribute__((ext_vector_type(8)))  __bf16 v8bf;
typedef __attribute__((ext_vector_type(2)))  __bf16 v2bf;
typedef __attribute__((ext_vector_type(8)))  float  v8f;
typedef __attribute__((ext_vector_type(4)))  float  v4f;
typedef __attribute__((ext_vector_type(2)))  float  v2f;

static constexpr int BSEQ = 4, LSEQ = 2048, NH = 16, HD = 64;
static constexpr int CC   = NH * HD;        // 1024
static constexpr int NTOK = BSEQ * LSEQ;    // 8192

union frag16 { v16bf v; v8bf h[2]; };

static __device__ __forceinline__ v8f wmma_bf16(const v16bf& a, const v16bf& b, const v8f& c) {
  return __builtin_amdgcn_wmma_f32_16x16x32_bf16(false, a, false, b, (short)0, c, false, false);
}

// Two contiguous 16-byte halves -> one 16-element fragment (2x ds/global _b128).
static __device__ __forceinline__ v16bf ld_frag(const __bf16* p0, const __bf16* p1) {
  frag16 f;
  f.h[0] = *(const v8bf*)p0;
  f.h[1] = *(const v8bf*)p1;
  return f.v;
}

// 8 contiguous f32 -> packed v8bf (global_load_b128 x2 + pk converts).
static __device__ __forceinline__ v8bf cvt8(const float* g) {
  v4f a = *(const v4f*)g;
  v4f b = *(const v4f*)(g + 4);
  v8bf r;
  #pragma unroll
  for (int j = 0; j < 4; ++j) { r[j] = (__bf16)a[j]; r[4 + j] = (__bf16)b[j]; }
  return r;
}

// gfx1250 async copy: 16 bytes global -> LDS, tracked by ASYNCcnt.
static __device__ __forceinline__ void async_copy_b128(void* lds, const void* gaddr) {
  unsigned lds_off = (unsigned)(unsigned long long)lds;   // addr[31:0] = LDS offset
  asm volatile("global_load_async_to_lds_b128 %0, %1, off"
               :: "v"(lds_off), "v"(gaddr) : "memory");
}
static __device__ __forceinline__ void wait_asynccnt0() {
  asm volatile("s_wait_asynccnt 0x0" ::: "memory");
}

// ---------------------------------------------------------------------------
// GEMM: C[M,N] = A[M,K] f32 @ B[K,N] f32 + bias[N].
// Block tile 64x128, BK=64, 8 waves, wave tile 32x32 (4 accums).
// Double-buffered LDS: As [2][64][64] (row-major), BsT [2][128][64]
// (transposed) -> every fragment is 2x ds_load_b128; one barrier per K-step.
// ---------------------------------------------------------------------------
__global__ __launch_bounds__(256) void gemm_bf16_wmma_kernel(
    const float* __restrict__ A, const float* __restrict__ B,
    const float* __restrict__ bias, float* __restrict__ C,
    int Mn, int Nn, int Kn)
{
  (void)Mn;
  constexpr int BM = 64, BN = 128, BK = 64;
  __shared__ __bf16 As[2][BM][BK];    // 16 KB
  __shared__ __bf16 BsT[2][BN][BK];   // 32 KB

  const int tid  = threadIdx.x;
  const int wave = tid >> 5;
  const int lane = tid & 31;
  const int bm   = blockIdx.y * BM;
  const int bn   = blockIdx.x * BN;
  const int wm   = (wave & 1) * 32;
  const int wn   = (wave >> 1) * 32;

  const int row  = lane & 15;
  const int kb8  = (lane >> 4) * 8;
  const int kr16 = (lane >> 4) * 16;
  const int rb8  = (lane >> 4) * 8;

  auto stage = [&](int buf, int k0) {
    #pragma unroll
    for (int i = 0; i < 2; ++i) {
      int u = tid + i * 256;                  // 512 units of 8 along K
      int r = u >> 3, ko = (u & 7) * 8;
      *(v8bf*)&As[buf][r][ko] = cvt8(&A[(size_t)(bm + r) * Kn + k0 + ko]);
    }
    #pragma unroll
    for (int i = 0; i < 4; ++i) {
      int u = tid + i * 256;                  // 1024 units
      int n = u & 127, ko = (u >> 7) * 8;
      v8bf pk;
      #pragma unroll
      for (int j = 0; j < 8; ++j)
        pk[j] = (__bf16)B[(size_t)(k0 + ko + j) * Nn + bn + n];
      *(v8bf*)&BsT[buf][n][ko] = pk;
    }
  };

  v8f acc[2][2] = {};

  stage(0, 0);
  __syncthreads();

  const int nk = Kn / BK;
  for (int kt = 0; kt < nk; ++kt) {
    const int cur = kt & 1;
    if (kt + 1 < nk) stage(cur ^ 1, (kt + 1) * BK);

    #pragma unroll
    for (int c = 0; c < 2; ++c) {             // two K=32 chunks
      v16bf fa[2], fb[2];
      #pragma unroll
      for (int i = 0; i < 2; ++i)
        fa[i] = ld_frag(&As[cur][wm + i * 16 + row][c * 32 + kb8],
                        &As[cur][wm + i * 16 + row][c * 32 + kb8 + 16]);
      #pragma unroll
      for (int j = 0; j < 2; ++j)
        fb[j] = ld_frag(&BsT[cur][wn + j * 16 + row][c * 32 + kb8],
                        &BsT[cur][wn + j * 16 + row][c * 32 + kb8 + 16]);
      #pragma unroll
      for (int i = 0; i < 2; ++i)
        #pragma unroll
        for (int j = 0; j < 2; ++j)
          acc[i][j] = wmma_bf16(fa[i], fb[j], acc[i][j]);
    }
    __syncthreads();
  }

  #pragma unroll
  for (int i = 0; i < 2; ++i)
    #pragma unroll
    for (int j = 0; j < 2; ++j) {
      int c0 = bn + wn + j * 16 + row;
      float bv = bias[c0];
      #pragma unroll
      for (int r = 0; r < 8; ++r) {
        int grow = bm + wm + i * 16 + rb8 + r;
        C[(size_t)grow * Nn + c0] = acc[i][j][r] + bv;
      }
    }
}

// ---------------------------------------------------------------------------
// QK-RMSNorm + repack: qkv[N][3C] f32 -> Q/K/V [B][H][L][D] bf16
// ---------------------------------------------------------------------------
__global__ __launch_bounds__(256) void rmsnorm_pack_kernel(
    const float* __restrict__ qkv, const float* __restrict__ gq,
    const float* __restrict__ gk,
    __bf16* __restrict__ Qo, __bf16* __restrict__ Ko, __bf16* __restrict__ Vo)
{
  int gw   = (blockIdx.x * blockDim.x + threadIdx.x) >> 5;
  int lane = threadIdx.x & 31;
  if (gw >= NTOK * NH) return;
  int token = gw / NH, head = gw % NH;
  int b = token / LSEQ, l = token % LSEQ;
  const float* base = qkv + (size_t)token * (3 * CC);
  size_t oidx = (((size_t)b * NH + head) * LSEQ + l) * HD;
  int d0 = lane * 2;

  v2f q = *(const v2f*)&base[head * HD + d0];
  v2f k = *(const v2f*)&base[CC + head * HD + d0];
  v2f v = *(const v2f*)&base[2 * CC + head * HD + d0];
  v2f gqv = *(const v2f*)&gq[head * HD + d0];
  v2f gkv = *(const v2f*)&gk[head * HD + d0];

  float sq = q[0] * q[0] + q[1] * q[1];
  float sk = k[0] * k[0] + k[1] * k[1];
  #pragma unroll
  for (int m = 16; m >= 1; m >>= 1) {
    sq += __shfl_xor(sq, m, 32);
    sk += __shfl_xor(sk, m, 32);
  }
  float qs = 8.0f / fmaxf(sqrtf(sq), 1e-12f);
  float ks = 8.0f / fmaxf(sqrtf(sk), 1e-12f);

  v2bf qo, ko, vo;
  qo[0] = (__bf16)(q[0] * qs * gqv[0]); qo[1] = (__bf16)(q[1] * qs * gqv[1]);
  ko[0] = (__bf16)(k[0] * ks * gkv[0]); ko[1] = (__bf16)(k[1] * ks * gkv[1]);
  vo[0] = (__bf16)v[0];                 vo[1] = (__bf16)v[1];
  *(v2bf*)&Qo[oidx + d0] = qo;
  *(v2bf*)&Ko[oidx + d0] = ko;
  *(v2bf*)&Vo[oidx + d0] = vo;
}

// ---------------------------------------------------------------------------
// Flash attention: block = (b,h, 64-query tile); 4 waves x 16 queries.
// Double-buffered tiles; K staged with global_load_async_to_lds_b128
// (ASYNCcnt), V transposed manually, psh per-wave P reshape.
// ---------------------------------------------------------------------------
__global__ __launch_bounds__(128) void flash_attn_kernel(
    const __bf16* __restrict__ Q, const __bf16* __restrict__ Kb,
    const __bf16* __restrict__ Vb, float* __restrict__ Hout)
{
  constexpr int LKB = 64;
  __shared__ __bf16 Ks[2][LKB][HD];     // 16 KB  [key][d]
  __shared__ __bf16 VsT[2][HD][LKB];    // 16 KB  [d][key]
  __shared__ __bf16 psh[4][16][LKB];    //  8 KB  per-wave P staging

  const int tid  = threadIdx.x;
  const int wave = tid >> 5;
  const int lane = tid & 31;
  const int qblocks = LSEQ / 64;
  const int bh = blockIdx.x / qblocks;
  const int qb = (blockIdx.x % qblocks) * 64;
  const int b  = bh / NH, h = bh % NH;

  const int row  = lane & 15;
  const int kb8  = (lane >> 4) * 8;
  const int kr16 = (lane >> 4) * 16;
  const int rb8  = (lane >> 4) * 8;

  const __bf16* kbase = Kb + (size_t)bh * LSEQ * HD;
  const __bf16* vbase = Vb + (size_t)bh * LSEQ * HD;

  // K tile: async DMA copy global -> LDS (no VGPR round trip)
  auto stageK = [&](int buf, int k0) {
    #pragma unroll
    for (int i = 0; i < 4; ++i) {
      int u = tid + i * 128;                  // 512 units of 8 bf16
      int key = u >> 3, ko = (u & 7) * 8;
      async_copy_b128(&Ks[buf][key][ko], &kbase[(size_t)(k0 + key) * HD + ko]);
    }
  };
  // V tile transposed: coalesced strided b16 loads + one b128 store per unit
  auto stageV = [&](int buf, int k0) {
    #pragma unroll
    for (int i = 0; i < 4; ++i) {
      int u = tid + i * 128;                  // 512 units
      int d = u & 63, ko = (u >> 6) * 8;
      v8bf pk;
      #pragma unroll
      for (int j = 0; j < 8; ++j)
        pk[j] = vbase[(size_t)(k0 + ko + j) * HD + d];
      *(v8bf*)&VsT[buf][d][ko] = pk;
    }
  };

  // Q fragments (16 queries x 64 d): two A-frags straight from global
  const __bf16* qptr = Q + (((size_t)bh * LSEQ) + qb + wave * 16) * HD;
  v16bf qf[2];
  #pragma unroll
  for (int hk = 0; hk < 2; ++hk)
    qf[hk] = ld_frag(qptr + row * HD + hk * 32 + kb8,
                     qptr + row * HD + hk * 32 + kb8 + 16);

  float mrow[8], lrow[8];
  #pragma unroll
  for (int r = 0; r < 8; ++r) { mrow[r] = -INFINITY; lrow[r] = 0.0f; }
  v8f o[4] = {};

  stageK(0, 0);
  stageV(0, 0);
  wait_asynccnt0();
  __syncthreads();

  const int nkb = LSEQ / LKB;
  for (int kt = 0; kt < nkb; ++kt) {
    const int cur = kt & 1;
    if (kt + 1 < nkb) {
      stageK(cur ^ 1, (kt + 1) * LKB);
      stageV(cur ^ 1, (kt + 1) * LKB);
    }

    // S = Q K^T : 4 N-tiles of 16 keys, K-dim = 64 (2 chunks)
    v8f s[4];
    #pragma unroll
    for (int nt = 0; nt < 4; ++nt) {
      v8f acc = {};
      #pragma unroll
      for (int hk = 0; hk < 2; ++hk) {
        v16bf bf = ld_frag(&Ks[cur][nt * 16 + row][hk * 32 + kb8],
                           &Ks[cur][nt * 16 + row][hk * 32 + kb8 + 16]);
        acc = wmma_bf16(qf[hk], bf, acc);
      }
      s[nt] = acc;
    }

    // online softmax (row M = rb8+r lives in one 16-lane half)
    #pragma unroll
    for (int r = 0; r < 8; ++r) {
      float lm = -INFINITY;
      #pragma unroll
      for (int nt = 0; nt < 4; ++nt) lm = fmaxf(lm, s[nt][r]);
      lm *= 0.125f;                           // 1/sqrt(D)
      #pragma unroll
      for (int m = 8; m >= 1; m >>= 1) lm = fmaxf(lm, __shfl_xor(lm, m, 32));
      float nm   = fmaxf(mrow[r], lm);
      float corr = __expf(mrow[r] - nm);
      float ls = 0.0f;
      #pragma unroll
      for (int nt = 0; nt < 4; ++nt) {
        float p = __expf(s[nt][r] * 0.125f - nm);
        s[nt][r] = p;
        ls += p;
      }
      #pragma unroll
      for (int m = 8; m >= 1; m >>= 1) ls += __shfl_xor(ls, m, 32);
      lrow[r] = lrow[r] * corr + ls;
      mrow[r] = nm;
      #pragma unroll
      for (int nt = 0; nt < 4; ++nt) o[nt][r] *= corr;
    }

    // C-layout -> A-layout reshape of P (per-wave; LDS in-order within wave)
    #pragma unroll
    for (int r = 0; r < 8; ++r)
      #pragma unroll
      for (int nt = 0; nt < 4; ++nt)
        psh[wave][rb8 + r][nt * 16 + row] = (__bf16)s[nt][r];
    __builtin_amdgcn_wave_barrier();

    v16bf pf[2];
    #pragma unroll
    for (int hk = 0; hk < 2; ++hk)
      pf[hk] = ld_frag(&psh[wave][row][hk * 32 + kb8],
                       &psh[wave][row][hk * 32 + kb8 + 16]);

    // O += P V : B-matrix from transposed V tile (contiguous key runs)
    #pragma unroll
    for (int nt = 0; nt < 4; ++nt)
      #pragma unroll
      for (int hk = 0; hk < 2; ++hk) {
        v16bf bv = ld_frag(&VsT[cur][nt * 16 + row][hk * 32 + kb8],
                           &VsT[cur][nt * 16 + row][hk * 32 + kb8 + 16]);
        o[nt] = wmma_bf16(pf[hk], bv, o[nt]);
      }

    wait_asynccnt0();                         // async writes to buf^1 landed
    __syncthreads();
  }

  #pragma unroll
  for (int r = 0; r < 8; ++r) {
    float invl = 1.0f / lrow[r];
    int token  = b * LSEQ + qb + wave * 16 + rb8 + r;
    #pragma unroll
    for (int nt = 0; nt < 4; ++nt)
      Hout[(size_t)token * CC + h * HD + nt * 16 + row] = o[nt][r] * invl;
  }
}

// ---------------------------------------------------------------------------
extern "C" void kernel_launch(void* const* d_in, const int* in_sizes, int n_in,
                              void* d_out, int out_size, void* d_ws, size_t ws_size,
                              hipStream_t stream) {
  (void)in_sizes; (void)n_in; (void)out_size; (void)ws_size;

  const float* x    = (const float*)d_in[0];   // [N, C]
  const float* Wqkv = (const float*)d_in[1];   // [C, 3C]
  const float* bqkv = (const float*)d_in[2];   // [3C]
  const float* gq   = (const float*)d_in[3];   // [H, D]
  const float* gk   = (const float*)d_in[4];   // [H, D]
  const float* Wout = (const float*)d_in[5];   // [C, C]
  const float* bout = (const float*)d_in[6];   // [C]
  float* out = (float*)d_out;                  // [N, C]

  char* p = (char*)d_ws;
  float*  qkv  = (float*)p;  p += (size_t)NTOK * 3 * CC * sizeof(float);   // 96 MB
  __bf16* Qb   = (__bf16*)p; p += (size_t)NTOK * CC * sizeof(__bf16);      // 16 MB
  __bf16* Kbuf = (__bf16*)p; p += (size_t)NTOK * CC * sizeof(__bf16);      // 16 MB
  __bf16* Vbuf = (__bf16*)p; p += (size_t)NTOK * CC * sizeof(__bf16);      // 16 MB
  float*  Hbuf = (float*)p;                                                // 32 MB

  // 1) qkv = x @ W_qkv + b_qkv
  dim3 g1((3 * CC) / 128, NTOK / 64);
  gemm_bf16_wmma_kernel<<<g1, 256, 0, stream>>>(x, Wqkv, bqkv, qkv, NTOK, 3 * CC, CC);

  // 2) RMSNorm(q,k) + pack q/k/v to bf16 [B,H,L,D]
  int nwaves = NTOK * NH;
  rmsnorm_pack_kernel<<<nwaves / 8, 256, 0, stream>>>(qkv, gq, gk, Qb, Kbuf, Vbuf);

  // 3) flash attention -> Hbuf [N, C] f32
  flash_attn_kernel<<<BSEQ * NH * (LSEQ / 64), 128, 0, stream>>>(Qb, Kbuf, Vbuf, Hbuf);

  // 4) out = H @ W_out + b_out
  dim3 g2(CC / 128, NTOK / 64);
  gemm_bf16_wmma_kernel<<<g2, 256, 0, stream>>>(Hbuf, Wout, bout, out, NTOK, CC, CC);
}
